// _SelfAttention_77953656423081
// MI455X (gfx1250) — compile-verified, matrix operand order checked by reading, not run on hardware
//
#include <hip/hip_runtime.h>
#include <hip/hip_bf16.h>

// ---------------------------------------------------------------------------
// Causal multi-head self-attention, MI455X (gfx1250), wave32 + WMMA bf16.
//   B=4, T=2048, C=1024, H=16, D=64
// Pipeline:
//   1) cvt fp32 -> bf16: x row-major; W_qkv/W_out transposed so GEMM B-tiles
//      are row-major (enables pure b128 async staging, no LDS transpose)
//   2) QKV GEMM (bf16 WMMA, f32 acc, async-to-LDS double buffering)
//        -> q[B,H,T,D], k[B,H,T,D], vT[B,H,D,T]
//   3) flash attention (online softmax), WMMA for QK^T and PV -> attn[B,T,C]
//   4) out-proj GEMM + bias -> d_out fp32
// ---------------------------------------------------------------------------

typedef unsigned short u16;
typedef __attribute__((ext_vector_type(16))) __bf16 v16bf;
typedef __attribute__((ext_vector_type(8)))  float  v8f;
typedef __attribute__((ext_vector_type(4)))  int    v4i;

#define T_SEQ   2048
#define C_DIM   1024
#define H_NUM   16
#define D_HEAD  64
#define LDSS    40          // padded LDS row stride (elements) for 32-wide tiles
#define TILE_E  (128 * LDSS)

#define AS1 __attribute__((address_space(1)))
#define AS3 __attribute__((address_space(3)))

union FragB16 {             // one WMMA 16-bit A/B fragment: 16 bf16 per lane
  v16bf  v;
  float4 f4[2];
  u16    u[16];
};

__device__ __forceinline__ u16 f2bf(float f) {
  unsigned u = __float_as_uint(f);
  u += 0x7FFFu + ((u >> 16) & 1u);      // round-to-nearest-even
  return (u16)(u >> 16);
}

__device__ __forceinline__ v8f v8f_zero() {
  v8f z;
#pragma unroll
  for (int i = 0; i < 8; ++i) z[i] = 0.0f;
  return z;
}

__device__ __forceinline__ v8f wmma_bf16(const FragB16& a, const FragB16& b, v8f c) {
  return __builtin_amdgcn_wmma_f32_16x16x32_bf16(
      false, a.v, false, b.v, (short)0, c, false, false);
}

// 16B global -> LDS copy; async (ASYNCcnt) when the toolchain exposes it.
__device__ __forceinline__ void cp_async_b128(u16* lptr, const u16* gptr) {
#if __has_builtin(__builtin_amdgcn_global_load_async_to_lds_b128)
  __builtin_amdgcn_global_load_async_to_lds_b128(
      (AS1 v4i*)gptr, (AS3 v4i*)lptr, 0, 0);
#else
  *(float4*)lptr = *(const float4*)gptr;
#endif
}

__device__ __forceinline__ void wait_async_all() {
#if __has_builtin(__builtin_amdgcn_s_wait_asynccnt)
  __builtin_amdgcn_s_wait_asynccnt(0);
#elif __has_builtin(__builtin_amdgcn_global_load_async_to_lds_b128)
  asm volatile("s_wait_asynccnt 0x0" ::: "memory");
#endif
}

// ------------------------------ fp32 -> bf16 -------------------------------
__global__ void k_cvt_bf16(const float* __restrict__ src, u16* __restrict__ dst, int n) {
  int i = blockIdx.x * blockDim.x + threadIdx.x;
  int stride = gridDim.x * blockDim.x;
  for (; i < n; i += stride) dst[i] = f2bf(src[i]);
}

// src [K][N] fp32 -> dst [N][K] bf16 (transposed weight for row-major B tiles)
__global__ void k_cvt_bf16_t(const float* __restrict__ src, u16* __restrict__ dst,
                             int K, int N) {
  int i = blockIdx.x * blockDim.x + threadIdx.x;
  int total = K * N;
  int stride = gridDim.x * blockDim.x;
  for (; i < total; i += stride) {
    int kx = i / N, nx = i - kx * N;        // coalesced read
    dst[(size_t)nx * K + kx] = f2bf(src[i]);
  }
}

// --------------------------- shared GEMM core ------------------------------
// 256 threads = 8 waves. Block tile 128x128, K-step 32, double-buffered LDS
// staged with async-to-LDS b128. Wave grid 4(m) x 2(n): 2x4 WMMA tiles/wave.
__device__ __forceinline__ void stage_tile(const u16* __restrict__ G, int ld,
                                           int row0, int k0, u16* buf, int tid) {
#pragma unroll
  for (int i = 0; i < 2; ++i) {
    int c    = tid + (i << 8);
    int row  = c >> 2;
    int col8 = (c & 3) << 3;
    cp_async_b128(buf + row * LDSS + col8,
                  G + (size_t)(row0 + row) * ld + (k0 + col8));
  }
}

__device__ __forceinline__ void gemm_core(
    const u16* __restrict__ A, const u16* __restrict__ Bt,
    int lda, int ldbt, int K, int m0, int n0,
    u16* ldsA, u16* ldsB, v8f acc[2][4])          // ldsA/ldsB: 2 buffers each
{
  const int tid  = threadIdx.x;
  const int lane = tid & 31;
  const int w    = tid >> 5;
  const int wm   = w >> 1;       // 0..3 -> 32-row slab
  const int wn   = w & 1;        // 0..1 -> 64-col slab
  const int lr   = lane & 15;
  const int lh   = lane >> 4;

#pragma unroll
  for (int mt = 0; mt < 2; ++mt)
#pragma unroll
    for (int nt = 0; nt < 4; ++nt) acc[mt][nt] = v8f_zero();

  const int kIters = K >> 5;

  stage_tile(A,  lda,  m0, 0, ldsA, tid);
  stage_tile(Bt, ldbt, n0, 0, ldsB, tid);
  wait_async_all();
  __syncthreads();

  for (int kk = 0; kk < kIters; ++kk) {
    const int cur = (kk & 1) ? TILE_E : 0;
    const int nxt = (kk & 1) ? 0 : TILE_E;
    if (kk + 1 < kIters) {                 // overlap: fill next buffer
      stage_tile(A,  lda,  m0, (kk + 1) << 5, ldsA + nxt, tid);
      stage_tile(Bt, ldbt, n0, (kk + 1) << 5, ldsB + nxt, tid);
    }

    FragB16 af[2], bf[4];
#pragma unroll
    for (int mt = 0; mt < 2; ++mt) {
      int row = wm * 32 + mt * 16 + lr;
      af[mt].f4[0] = *(const float4*)(ldsA + cur + row * LDSS + lh * 8);
      af[mt].f4[1] = *(const float4*)(ldsA + cur + row * LDSS + 16 + lh * 8);
    }
#pragma unroll
    for (int nt = 0; nt < 4; ++nt) {
      int nn = wn * 64 + nt * 16 + lr;
      bf[nt].f4[0] = *(const float4*)(ldsB + cur + nn * LDSS + lh * 8);
      bf[nt].f4[1] = *(const float4*)(ldsB + cur + nn * LDSS + 16 + lh * 8);
    }
#pragma unroll
    for (int mt = 0; mt < 2; ++mt)
#pragma unroll
      for (int nt = 0; nt < 4; ++nt)
        acc[mt][nt] = wmma_bf16(af[mt], bf[nt], acc[mt][nt]);

    wait_async_all();                      // next buffer landed
    __syncthreads();                       // everyone done reading cur buffer
  }
}

// ----------------------- QKV projection + scatter --------------------------
__global__ __launch_bounds__(256) void k_gemm_qkv(
    const u16* __restrict__ xb, const u16* __restrict__ wbt,
    const float* __restrict__ bias,
    u16* __restrict__ qo, u16* __restrict__ ko, u16* __restrict__ vto)
{
  __shared__ __align__(16) u16 ldsA[2 * TILE_E];
  __shared__ __align__(16) u16 ldsB[2 * TILE_E];
  v8f acc[2][4];
  const int m0 = blockIdx.y * 128;
  const int n0 = blockIdx.x * 128;
  gemm_core(xb, wbt, C_DIM, C_DIM, C_DIM, m0, n0, ldsA, ldsB, acc);

  const int lane = threadIdx.x & 31;
  const int w    = threadIdx.x >> 5;
  const int wm = w >> 1, wn = w & 1, lr = lane & 15, lh = lane >> 4;
#pragma unroll
  for (int mt = 0; mt < 2; ++mt)
#pragma unroll
    for (int nt = 0; nt < 4; ++nt)
#pragma unroll
      for (int r = 0; r < 8; ++r) {
        int m = m0 + wm * 32 + mt * 16 + r + 8 * lh;
        int n = n0 + wn * 64 + nt * 16 + lr;
        u16 bv = f2bf(acc[mt][nt][r] + bias[n]);
        int b = m >> 11, t = m & (T_SEQ - 1);
        int part = n >> 10, rem = n & (C_DIM - 1);
        int h = rem >> 6, d = rem & (D_HEAD - 1);
        size_t bh = (size_t)b * H_NUM + h;
        if (part == 0)       qo [(bh * T_SEQ + t) * D_HEAD + d] = bv;
        else if (part == 1)  ko [(bh * T_SEQ + t) * D_HEAD + d] = bv;
        else                 vto[(bh * D_HEAD + d) * T_SEQ + t] = bv;   // V^T
      }
}

// --------------------------- flash attention -------------------------------
// grid: (T/128, B*H). 8 waves/block, each wave owns 16 query rows.
__global__ __launch_bounds__(256) void k_attn(
    const u16* __restrict__ q, const u16* __restrict__ k,
    const u16* __restrict__ vt, u16* __restrict__ o)
{
  __shared__ __align__(16) u16 ldsP[8 * 16 * LDSS];   // per-wave 16x32 P tile
  const int w    = threadIdx.x >> 5;
  const int lane = threadIdx.x & 31;
  const int lr = lane & 15, lh = lane >> 4;
  const int bh = blockIdx.y;
  const int qs = blockIdx.x * 128 + w * 16;           // this wave's first q row
  const size_t qkbase = (size_t)bh * T_SEQ * D_HEAD;
  const size_t vtbase = (size_t)bh * D_HEAD * T_SEQ;
  u16* P = ldsP + w * 16 * LDSS;

  // Q fragments (16 rows x 64 -> two K=32 A-fragments), loaded once
  FragB16 qf[2];
  const u16* qrow = q + qkbase + (size_t)(qs + lr) * D_HEAD;
  qf[0].f4[0] = *(const float4*)(qrow +      lh * 8);
  qf[0].f4[1] = *(const float4*)(qrow + 16 + lh * 8);
  qf[1].f4[0] = *(const float4*)(qrow + 32 + lh * 8);
  qf[1].f4[1] = *(const float4*)(qrow + 48 + lh * 8);

  v8f accO[4];
#pragma unroll
  for (int nt = 0; nt < 4; ++nt) accO[nt] = v8f_zero();
  float mrow[8], lrow[8];
#pragma unroll
  for (int r = 0; r < 8; ++r) { mrow[r] = -1e30f; lrow[r] = 0.0f; }

  const int nChunks = (qs + 16 + 31) >> 5;            // causal: kv <= qs+15
  for (int kc = 0; kc < nChunks; ++kc) {
    const int kv0 = kc << 5;

    if (kc + 1 < nChunks) {                           // near-prefetch next chunk
      __builtin_prefetch(k  + qkbase + (size_t)(kv0 + 32 + lane) * D_HEAD, 0, 3);
      __builtin_prefetch(vt + vtbase + (size_t)(lane * 2) * T_SEQ + kv0 + 32, 0, 3);
    }

    // K fragments: two 16-key tiles x two d-halves
    FragB16 kf[2][2];
#pragma unroll
    for (int t2 = 0; t2 < 2; ++t2) {
      const u16* krow = k + qkbase + (size_t)(kv0 + t2 * 16 + lr) * D_HEAD;
      kf[t2][0].f4[0] = *(const float4*)(krow +      lh * 8);
      kf[t2][0].f4[1] = *(const float4*)(krow + 16 + lh * 8);
      kf[t2][1].f4[0] = *(const float4*)(krow + 32 + lh * 8);
      kf[t2][1].f4[1] = *(const float4*)(krow + 48 + lh * 8);
    }

    // S = Q K^T for 16x32 score tile (two 16x16 WMMA columns, K=32 each half)
    v8f s[2];
#pragma unroll
    for (int t2 = 0; t2 < 2; ++t2) {
      v8f z = v8f_zero();
      z = wmma_bf16(qf[0], kf[t2][0], z);
      z = wmma_bf16(qf[1], kf[t2][1], z);
      s[t2] = z;
    }

    // scale + causal mask + online softmax (row lives in one 16-lane half)
#pragma unroll
    for (int r = 0; r < 8; ++r) {
      const int qi = qs + r + 8 * lh;
      float a0 = s[0][r] * 0.125f;                    // 1/sqrt(64)
      float a1 = s[1][r] * 0.125f;
      if (kv0 + lr      > qi) a0 = -1e30f;
      if (kv0 + 16 + lr > qi) a1 = -1e30f;

      float mx = fmaxf(a0, a1);
      mx = fmaxf(mx, __shfl_xor(mx, 8, 32));
      mx = fmaxf(mx, __shfl_xor(mx, 4, 32));
      mx = fmaxf(mx, __shfl_xor(mx, 2, 32));
      mx = fmaxf(mx, __shfl_xor(mx, 1, 32));
      const float mnew  = fmaxf(mrow[r], mx);
      const float alpha = __expf(mrow[r] - mnew);
      const float e0 = __expf(a0 - mnew);
      const float e1 = __expf(a1 - mnew);
      float sum = e0 + e1;
      sum += __shfl_xor(sum, 8, 32);
      sum += __shfl_xor(sum, 4, 32);
      sum += __shfl_xor(sum, 2, 32);
      sum += __shfl_xor(sum, 1, 32);
      lrow[r] = lrow[r] * alpha + sum;
      mrow[r] = mnew;
#pragma unroll
      for (int nt = 0; nt < 4; ++nt) accO[nt][r] *= alpha;

      // C-layout -> LDS (bf16) so we can reload P in A-fragment layout
      P[(r + 8 * lh) * LDSS + lr]      = f2bf(e0);
      P[(r + 8 * lh) * LDSS + 16 + lr] = f2bf(e1);
    }

    FragB16 pf;                                        // P as 16x32 A-fragment
    pf.f4[0] = *(const float4*)(P + lr * LDSS + lh * 8);
    pf.f4[1] = *(const float4*)(P + lr * LDSS + 16 + lh * 8);

    // O += P @ V  (V stored transposed -> contiguous B-fragment loads)
#pragma unroll
    for (int nt = 0; nt < 4; ++nt) {
      FragB16 vf;
      const u16* vrow = vt + vtbase + (size_t)(nt * 16 + lr) * T_SEQ + kv0;
      vf.f4[0] = *(const float4*)(vrow +      lh * 8);
      vf.f4[1] = *(const float4*)(vrow + 16 + lh * 8);
      accO[nt] = wmma_bf16(pf, vf, accO[nt]);
    }
  }

  // normalize + store attn output as [B,T,C] bf16
  const int b = bh >> 4, hh = bh & (H_NUM - 1);
#pragma unroll
  for (int r = 0; r < 8; ++r) {
    const int t = qs + r + 8 * lh;
    const float inv = 1.0f / lrow[r];
#pragma unroll
    for (int nt = 0; nt < 4; ++nt) {
      const int d = nt * 16 + lr;
      o[((size_t)(b * T_SEQ + t)) * C_DIM + hh * D_HEAD + d] = f2bf(accO[nt][r] * inv);
    }
  }
}

// ------------------------- output projection -------------------------------
__global__ __launch_bounds__(256) void k_gemm_out(
    const u16* __restrict__ ab, const u16* __restrict__ wbt,
    const float* __restrict__ bias, float* __restrict__ out)
{
  __shared__ __align__(16) u16 ldsA[2 * TILE_E];
  __shared__ __align__(16) u16 ldsB[2 * TILE_E];
  v8f acc[2][4];
  const int m0 = blockIdx.y * 128;
  const int n0 = blockIdx.x * 128;
  gemm_core(ab, wbt, C_DIM, C_DIM, C_DIM, m0, n0, ldsA, ldsB, acc);

  const int lane = threadIdx.x & 31;
  const int w    = threadIdx.x >> 5;
  const int wm = w >> 1, wn = w & 1, lr = lane & 15, lh = lane >> 4;
#pragma unroll
  for (int mt = 0; mt < 2; ++mt)
#pragma unroll
    for (int nt = 0; nt < 4; ++nt)
#pragma unroll
      for (int r = 0; r < 8; ++r) {
        int m = m0 + wm * 32 + mt * 16 + r + 8 * lh;
        int n = n0 + wn * 64 + nt * 16 + lr;
        out[(size_t)m * C_DIM + n] = acc[mt][nt][r] + bias[n];
      }
}

// ------------------------------- launcher ----------------------------------
extern "C" void kernel_launch(void* const* d_in, const int* in_sizes, int n_in,
                              void* d_out, int out_size, void* d_ws, size_t ws_size,
                              hipStream_t stream) {
  const float* x      = (const float*)d_in[0];   // [4,2048,1024]
  const float* W_qkv  = (const float*)d_in[1];   // [1024,3072]
  const float* b_qkv  = (const float*)d_in[2];   // [3072]
  const float* W_out  = (const float*)d_in[3];   // [1024,1024]
  const float* b_out  = (const float*)d_in[4];   // [1024]
  float* out = (float*)d_out;

  const int NX  = 4 * T_SEQ * C_DIM;             // 8388608
  const int NWQ = C_DIM * 3 * C_DIM;             // 3145728
  const int NWO = C_DIM * C_DIM;                 // 1048576

  u16* ws     = (u16*)d_ws;
  u16* xb     = ws;
  u16* wqkvT  = xb    + NX;                      // [3C][C] bf16
  u16* woutT  = wqkvT + NWQ;                     // [C][C]  bf16
  u16* qws    = woutT + NWO;                     // [B,H,T,D]
  u16* kws    = qws   + NX;                      // [B,H,T,D]
  u16* vtws   = kws   + NX;                      // [B,H,D,T]
  u16* attnb  = vtws  + NX;                      // [B,T,C]

  k_cvt_bf16  <<<(NX  + 255) / 256, 256, 0, stream>>>(x, xb, NX);
  k_cvt_bf16_t<<<(NWQ + 255) / 256, 256, 0, stream>>>(W_qkv, wqkvT, C_DIM, 3 * C_DIM);
  k_cvt_bf16_t<<<(NWO + 255) / 256, 256, 0, stream>>>(W_out, woutT, C_DIM, C_DIM);

  k_gemm_qkv<<<dim3(3 * C_DIM / 128, NX / C_DIM / 128), 256, 0, stream>>>(
      xb, wqkvT, b_qkv, qws, kws, vtws);

  k_attn<<<dim3(T_SEQ / 128, 4 * H_NUM), 256, 0, stream>>>(qws, kws, vtws, attnb);

  k_gemm_out<<<dim3(C_DIM / 128, NX / C_DIM / 128), 256, 0, stream>>>(
      attnb, woutT, b_out, out);
}
